// RNN_67997922230777
// MI455X (gfx1250) — compile-verified, matrix operand order checked by reading, not run on hardware
//
#include <hip/hip_runtime.h>
#include <hip/hip_bf16.h>
#include <stdint.h>

// Problem dimensions (fixed by the reference)
#define T_STEPS 8192
#define OBS     256
#define IN_SZ   512      // 2*OBS
#define HID     2048
#define H3      6144     // 3*HID
#define OUT_SZ  256
#define NBLK_SCAN 128    // persistent scan grid (must be co-resident)

typedef __attribute__((ext_vector_type(16))) __bf16 v16bf;
typedef __attribute__((ext_vector_type(8)))  float  v8f;

// ---------------------------------------------------------------- helpers ---

__device__ __forceinline__ uint16_t f2bf(float f) {
    uint32_t u = __float_as_uint(f);
    uint32_t r = (u + 0x7fffu + ((u >> 16) & 1u)) >> 16;   // round-to-nearest-even
    return (uint16_t)r;
}

// Load a 16x32 bf16 A/B fragment's per-lane 16 elements.
// Per ISA 7.12.2: lane holds K = [kb..kb+7] and [kb+16..kb+23] of its row,
// where kb = 8*(lane>=16). Two 16B loads.
__device__ __forceinline__ v16bf load_frag(const uint16_t* __restrict__ p, int off) {
    union { uint4 u[2]; v16bf v; } f;
    f.u[0] = *(const uint4*)(p + off);
    f.u[1] = *(const uint4*)(p + off + 16);
    return f.v;
}

__device__ __forceinline__ v8f wmma_bf16(v16bf a, v16bf b, v8f c) {
    // (neg_a, A, neg_b, B, c_mod, C, reuse_a, reuse_b)
    return __builtin_amdgcn_wmma_f32_16x16x32_bf16(false, a, false, b, (short)0, c,
                                                   false, false);
}

__device__ __forceinline__ float sigmoidf_(float x) {
    return 1.0f / (1.0f + __expf(-x));
}

__device__ __forceinline__ void grid_sync(int* cnt, int* gen, int nblk) {
    __syncthreads();
    if (threadIdx.x == 0) {
        __threadfence();
        int g = __hip_atomic_load(gen, __ATOMIC_RELAXED, __HIP_MEMORY_SCOPE_AGENT);
        int arrived = __hip_atomic_fetch_add(cnt, 1, __ATOMIC_ACQ_REL,
                                             __HIP_MEMORY_SCOPE_AGENT);
        if (arrived == nblk - 1) {
            __hip_atomic_store(cnt, 0, __ATOMIC_RELEASE, __HIP_MEMORY_SCOPE_AGENT);
            __hip_atomic_store(gen, g + 1, __ATOMIC_RELEASE, __HIP_MEMORY_SCOPE_AGENT);
        } else {
            while (__hip_atomic_load(gen, __ATOMIC_ACQUIRE,
                                     __HIP_MEMORY_SCOPE_AGENT) == g) {
                __builtin_amdgcn_s_sleep(1);
            }
        }
    }
    __syncthreads();
}

// ---------------------------------------------------------- small kernels ---

__global__ void k_init(float* h32, uint16_t* hbuf, int* barrier) {
    int i = blockIdx.x * blockDim.x + threadIdx.x;
    if (i < HID) { h32[i] = 0.0f; hbuf[i] = 0; }
    if (i == 0)  { barrier[0] = 0; barrier[1] = 0; }
}

// Build x = [nan_to_num(ys), mask] as bf16, row-major [T, 512]
__global__ void k_build_x(const float* __restrict__ ys, uint16_t* __restrict__ xbf) {
    int i = blockIdx.x * blockDim.x + threadIdx.x;
    if (i >= T_STEPS * OBS) return;
    int t = i / OBS, c = i % OBS;
    float y = ys[i];
    bool isnan = (y != y);
    xbf[(size_t)t * IN_SZ + c]       = f2bf(isnan ? 0.0f : y);
    xbf[(size_t)t * IN_SZ + OBS + c] = f2bf(isnan ? 0.0f : 1.0f);
}

__global__ void k_cvt_bf16(const float* __restrict__ src, uint16_t* __restrict__ dst,
                           int n) {
    int i = blockIdx.x * blockDim.x + threadIdx.x;
    if (i < n) dst[i] = f2bf(src[i]);
}

// --------------------------------------------------------------- big GEMM ---
// C[M,N] = A[M,K](bf16,rm) * B[N,K](bf16,rm)^T + bias[N], fp32 out.
// One wave computes a 32x32 tile (4 wmma accumulators).
__global__ __launch_bounds__(256)
void k_wmma_gemm_bias(const uint16_t* __restrict__ A, const uint16_t* __restrict__ B,
                      const float* __restrict__ bias, float* __restrict__ C,
                      int M, int N, int K) {
    int ntn = N >> 5;
    int gw  = blockIdx.x * (blockDim.x >> 5) + (threadIdx.x >> 5);
    int tm  = gw / ntn, tn = gw % ntn;
    if (tm * 32 >= M) return;                // uniform per wave
    int lane = threadIdx.x & 31;
    int lrow = lane & 15;
    int kb   = (lane >> 4) << 3;

    const uint16_t* a0p = A + (size_t)(tm * 32 + lrow)      * K;
    const uint16_t* a1p = A + (size_t)(tm * 32 + 16 + lrow) * K;
    const uint16_t* b0p = B + (size_t)(tn * 32 + lrow)      * K;
    const uint16_t* b1p = B + (size_t)(tn * 32 + 16 + lrow) * K;

    v8f c00{}, c01{}, c10{}, c11{};
#pragma unroll 4
    for (int k0 = 0; k0 < K; k0 += 32) {
        v16bf a0 = load_frag(a0p, k0 + kb);
        v16bf a1 = load_frag(a1p, k0 + kb);
        v16bf b0 = load_frag(b0p, k0 + kb);
        v16bf b1 = load_frag(b1p, k0 + kb);
        c00 = wmma_bf16(a0, b0, c00);
        c01 = wmma_bf16(a0, b1, c01);
        c10 = wmma_bf16(a1, b0, c10);
        c11 = wmma_bf16(a1, b1, c11);
    }

    // D(m,n): lane = n + 16*(m>=8), vgpr r = m%8
    int nl = (lane < 16) ? lane : lane - 16;
    int mo = (lane < 16) ? 0 : 8;
#pragma unroll
    for (int i = 0; i < 2; ++i) {
#pragma unroll
        for (int j = 0; j < 2; ++j) {
            v8f acc = (i == 0) ? ((j == 0) ? c00 : c01) : ((j == 0) ? c10 : c11);
            int n  = tn * 32 + 16 * j + nl;
            float bv = bias[n];
            int mbase = tm * 32 + 16 * i + mo;
#pragma unroll
            for (int r = 0; r < 8; ++r)
                C[(size_t)(mbase + r) * N + n] = acc[r] + bv;
        }
    }
}

// ------------------------------------------------------- persistent scan ----
// 128 blocks x 96 threads (3 waves). Block b owns hidden units
// j in [16b, 16b+16). Wave w computes the 16-row wh tile of gate w via WMMA
// (h broadcast into B; D column 0 = wh_tile . h). Gates reduce through LDS;
// one device-wide barrier per timestep; h buffers ping-pong by t parity.
__global__ __launch_bounds__(96)
void k_scan(const uint16_t* __restrict__ whbf,    // [6144,2048] bf16
            const float*    __restrict__ igates,  // [T,6144] f32
            const float*    __restrict__ bn,      // [2048]
            float*          __restrict__ h32,     // [2][2048] f32 ping-pong
            uint16_t*       __restrict__ hbuf,    // [2][2048] bf16 ping-pong
            uint16_t*       __restrict__ hidden,  // [T,2048] bf16
            int*            __restrict__ barrier) // {count, generation}
{
    __shared__ float hg[48];                      // [gate 0..2][16]
    int tid  = threadIdx.x;
    int w    = tid >> 5;
    int lane = tid & 31;
    int jt   = blockIdx.x * 16;
    int lrow = lane & 15;
    int kb   = (lane >> 4) << 3;
    const uint16_t* ap = whbf + (size_t)(w * HID + jt + lrow) * HID;

    for (int t = 0; t < T_STEPS; ++t) {
        const uint16_t* hb = hbuf + (size_t)(t & 1) * HID;
        v8f acc{};
#pragma unroll 4
        for (int k0 = 0; k0 < HID; k0 += 32) {
            v16bf a = load_frag(ap, k0 + kb);
            v16bf b = load_frag(hb, k0 + kb);   // same h for every column
            acc = wmma_bf16(a, b, acc);
        }
        // column 0 of D: m<8 in lane 0, m>=8 in lane 16
        if (lane == 0) {
#pragma unroll
            for (int r = 0; r < 8; ++r) hg[w * 16 + r] = acc[r];
        }
        if (lane == 16) {
#pragma unroll
            for (int r = 0; r < 8; ++r) hg[w * 16 + 8 + r] = acc[r];
        }
        __syncthreads();

        if (tid < 16) {
            int j = jt + tid;
            float hr = hg[tid], hz = hg[16 + tid], hn = hg[32 + tid];
            const float* ig = igates + (size_t)t * H3;
            float r = sigmoidf_(ig[j] + hr);
            float z = sigmoidf_(ig[HID + j] + hz);
            float n = tanhf(ig[2 * HID + j] + r * (hn + bn[j]));
            float hold = h32[(size_t)(t & 1) * HID + j];
            float hnew = n + z * (hold - n);
            h32[(size_t)((t + 1) & 1) * HID + j] = hnew;
            uint16_t hb16 = f2bf(hnew);
            hbuf[(size_t)((t + 1) & 1) * HID + j] = hb16;
            hidden[(size_t)t * HID + j] = hb16;
        }
        grid_sync(barrier, barrier + 1, NBLK_SCAN);
    }
}

// ------------------------------------------------------------------ launch --

extern "C" void kernel_launch(void* const* d_in, const int* in_sizes, int n_in,
                              void* d_out, int out_size, void* d_ws, size_t ws_size,
                              hipStream_t stream) {
    // Inputs (setup_inputs order): ts, ys, wi, wh, b, bn, Wl, bl — all fp32
    const float* ys = (const float*)d_in[1];
    const float* wi = (const float*)d_in[2];
    const float* wh = (const float*)d_in[3];
    const float* b  = (const float*)d_in[4];
    const float* bn = (const float*)d_in[5];
    const float* Wl = (const float*)d_in[6];
    const float* bl = (const float*)d_in[7];
    float* out = (float*)d_out;

    // Workspace carve-up (all chunks 16B-aligned by construction)
    uint16_t* x_bf   = (uint16_t*)d_ws;                                  // [T,512]
    uint16_t* wi_bf  = x_bf  + (size_t)T_STEPS * IN_SZ;                  // [6144,512]
    uint16_t* wh_bf  = wi_bf + (size_t)H3 * IN_SZ;                       // [6144,2048]
    uint16_t* Wl_bf  = wh_bf + (size_t)H3 * HID;                         // [256,2048]
    float*    igates = (float*)(Wl_bf + (size_t)OUT_SZ * HID);           // [T,6144]
    uint16_t* hidden = (uint16_t*)(igates + (size_t)T_STEPS * H3);       // [T,2048]
    float*    h32    = (float*)(hidden + (size_t)T_STEPS * HID);         // [2,2048]
    uint16_t* hbuf   = (uint16_t*)(h32 + 2 * HID);                       // [2,2048]
    int*      barrier = (int*)(hbuf + 2 * HID);                          // {cnt, gen}

    // 1) deterministic state init (barrier + h0 = 0) — re-run on every call
    k_init<<<(HID + 255) / 256, 256, 0, stream>>>(h32, hbuf, barrier);

    // 2) build bf16 operands
    k_build_x<<<(T_STEPS * OBS + 255) / 256, 256, 0, stream>>>(ys, x_bf);
    k_cvt_bf16<<<(H3 * IN_SZ + 255) / 256, 256, 0, stream>>>(wi, wi_bf, H3 * IN_SZ);
    k_cvt_bf16<<<(H3 * HID + 255) / 256, 256, 0, stream>>>(wh, wh_bf, H3 * HID);
    k_cvt_bf16<<<(OUT_SZ * HID + 255) / 256, 256, 0, stream>>>(Wl, Wl_bf, OUT_SZ * HID);

    // 3) igates = x @ wi^T + b  : [8192,512]x[512,6144]
    {
        int waves = (T_STEPS / 32) * (H3 / 32);          // 49152
        k_wmma_gemm_bias<<<waves / 8, 256, 0, stream>>>(x_bf, wi_bf, b, igates,
                                                        T_STEPS, H3, IN_SZ);
    }

    // 4) sequential GRU scan (persistent kernel, one grid barrier per step)
    k_scan<<<NBLK_SCAN, 96, 0, stream>>>(wh_bf, igates, bn, h32, hbuf, hidden,
                                         barrier);

    // 5) out = hidden @ Wl^T + bl : [8192,2048]x[2048,256]
    {
        int waves = (T_STEPS / 32) * (OUT_SZ / 32);      // 2048
        k_wmma_gemm_bias<<<waves / 8, 256, 0, stream>>>(hidden, Wl_bf, bl, out,
                                                        T_STEPS, OUT_SZ, HID);
    }
}
